// EdgeImportanceRefiner_73478300500072
// MI455X (gfx1250) — matrix operand order repaired and checked by reading, not compiled, not run on hardware
//
#include <hip/hip_runtime.h>
#include <hip/hip_bf16.h>
#include <stdint.h>

typedef __attribute__((ext_vector_type(16))) _Float16 v16h;
typedef __attribute__((ext_vector_type(8)))  float    v8f;

#define D_IN 256
#define H1 128
#define H2 64
#define K1_CHUNKS 25          // 800 = 25 * 32 ; feature 800 (grad score) is a rank-1 update
#define K2_CHUNKS 4           // 128 = 4 * 32
#define BN_EPS 1e-5f
#define NBINS 64
#define L1_STRIDE_H 808       // halves per staged feature row (800 + 8 pad, conflict-free)
#define L1_STRIDE_D 404       // same in dwords

union H2U { _Float16 h[2]; uint32_t u; };
union AV  { uint32_t u[8]; uint4 q[2]; v16h v; };

// ---------------- stats zeroing ----------------
__global__ __launch_bounds__(256) void zero_stats(float* __restrict__ s1,
                                                  float* __restrict__ s2) {
  int i = blockIdx.x * 256 + threadIdx.x;
  if (i < NBINS * 2 * H1) s1[i] = 0.f;
  if (i < NBINS * 2 * H2) s2[i] = 0.f;
}

// ---------------- pre-swizzle W1 (801x128 f32) into WMMA B lane layout, f16 ----------------
// layout: [chunk(25)][ntile(8)][lane(32)][vgpr(8)] dwords; each dword = 2 consecutive-K f16
__global__ __launch_bounds__(256) void pack_w1(const float* __restrict__ W1,
                                               uint32_t* __restrict__ B1p) {
  int idx = blockIdx.x * 256 + threadIdx.x;
  if (idx >= K1_CHUNKS * 8 * 32 * 8) return;
  int j = idx & 7, lane = (idx >> 3) & 31, nt = (idx >> 8) & 7, c = idx >> 11;
  int n = (nt << 4) + (lane & 15);
  int k = (c << 5) + ((lane >> 4) << 4) + (j << 1);   // < 800
  H2U p;
  p.h[0] = (_Float16)W1[(size_t)k * H1 + n];
  p.h[1] = (_Float16)W1[(size_t)(k + 1) * H1 + n];
  B1p[idx] = p.u;
}

// ---------------- layer 1: gather + [32 x 800] @ [800 x 128] WMMA + ReLU + stats ----------------
__global__ __launch_bounds__(256) void layer1(
    const float* __restrict__ NE, const float* __restrict__ GS,
    const float* __restrict__ STEP, const float* __restrict__ W1,
    const float* __restrict__ B1, const long long* __restrict__ EI,
    const uint32_t* __restrict__ B1p, _Float16* __restrict__ X1,
    float* __restrict__ S1, int E) {
  __shared__ _Float16 feats[32 * L1_STRIDE_H];
  __shared__ _Float16 otile[32 * H1];
  __shared__ float gradv[32];

  const int tid = threadIdx.x;
  const int e0 = blockIdx.x * 32;

  // ---- stage 32 edges of 800 f16 features into LDS (8 threads per edge row) ----
  {
    int r = tid >> 3, t = tid & 7;
    int e = e0 + r;
    _Float16* frow = &feats[r * L1_STRIDE_H];
    if (e < E) {
      int src = (int)EI[e];
      int dst = (int)EI[(long long)E + e];
      const float4* hu4 = (const float4*)(NE + (long long)src * D_IN);
      const float4* hv4 = (const float4*)(NE + (long long)dst * D_IN);
      for (int q = t; q < D_IN / 4; q += 8) {
        float4 a = hu4[q], b = hv4[q];
        int k = q << 2;
        frow[k + 0] = (_Float16)a.x; frow[k + 1] = (_Float16)a.y;
        frow[k + 2] = (_Float16)a.z; frow[k + 3] = (_Float16)a.w;
        frow[256 + k + 0] = (_Float16)b.x; frow[256 + k + 1] = (_Float16)b.y;
        frow[256 + k + 2] = (_Float16)b.z; frow[256 + k + 3] = (_Float16)b.w;
        frow[512 + k + 0] = (_Float16)fabsf(a.x - b.x);
        frow[512 + k + 1] = (_Float16)fabsf(a.y - b.y);
        frow[512 + k + 2] = (_Float16)fabsf(a.z - b.z);
        frow[512 + k + 3] = (_Float16)fabsf(a.w - b.w);
      }
      for (int k = t; k < 32; k += 8) frow[768 + k] = (_Float16)STEP[k];
      if (t == 0) gradv[r] = GS[e];
    } else {
      for (int k = t; k < 800; k += 8) frow[k] = (_Float16)0.f;
      if (t == 0) gradv[r] = 0.f;
    }
  }
  __syncthreads();

  // ---- WMMA: each wave owns one 16-wide N tile, both 16-row M tiles (shared B) ----
  const int wave = tid >> 5, lane = tid & 31;
  const int nt = wave;                       // N tile 0..7 -> 128 outputs
  const int row = lane & 15;
  const int kbh = (lane >> 4) << 3;          // dword offset: 0 (lanes 0-15) / 8 (lanes 16-31)
  const uint32_t* fd = (const uint32_t*)feats;

  v8f acc0 = {}; v8f acc1 = {};
  for (int c = 0; c < K1_CHUNKS; ++c) {
    AV b;
    const uint4* bq = (const uint4*)(B1p + ((((c << 3) + nt) << 5) + lane) * 8);
    b.q[0] = bq[0]; b.q[1] = bq[1];
    if (c + 1 < K1_CHUNKS)
      __builtin_prefetch(B1p + (((((c + 1) << 3) + nt) << 5) + lane) * 8, 0, 1);
    AV a0, a1;
    int base0 = row * L1_STRIDE_D + (c << 4) + kbh;
    a0.q[0] = *(const uint4*)(fd + base0);
    a0.q[1] = *(const uint4*)(fd + base0 + 4);
    int base1 = base0 + 16 * L1_STRIDE_D;
    a1.q[0] = *(const uint4*)(fd + base1);
    a1.q[1] = *(const uint4*)(fd + base1 + 4);
    acc0 = __builtin_amdgcn_wmma_f32_16x16x32_f16(false, a0.v, false, b.v,
                                                  (short)0, acc0, false, false);
    acc1 = __builtin_amdgcn_wmma_f32_16x16x32_f16(false, a1.v, false, b.v,
                                                  (short)0, acc1, false, false);
  }

  // ---- rank-1 grad column + bias + ReLU + per-feature stats ----
  const int n = (lane & 15) + (nt << 4);
  const int mb = (lane >> 4) << 3;
  const float wlast = W1[(size_t)800 * H1 + n];
  const float bias = B1[n];
  float s = 0.f, s2 = 0.f;
  for (int i = 0; i < 8; ++i) {
    int m0 = mb + i, m1 = m0 + 16;
    float v0 = fmaxf(acc0[i] + gradv[m0] * wlast + bias, 0.f);
    float v1 = fmaxf(acc1[i] + gradv[m1] * wlast + bias, 0.f);
    otile[m0 * H1 + n] = (_Float16)v0;
    otile[m1 * H1 + n] = (_Float16)v1;
    if (e0 + m0 < E) { s += v0; s2 += v0 * v0; }
    if (e0 + m1 < E) { s += v1; s2 += v1 * v1; }
  }
  s  += __shfl_xor(s, 16, 32);
  s2 += __shfl_xor(s2, 16, 32);
  if (lane < 16) {
    float* bin = S1 + (size_t)(blockIdx.x & (NBINS - 1)) * (2 * H1);
    atomicAdd(bin + n, s);
    atomicAdd(bin + H1 + n, s2);
  }
  __syncthreads();

  // ---- coalesced b128 store of the 32x128 f16 tile ----
  const uint4* ot = (const uint4*)otile;
  for (int q = tid; q < 32 * H1 / 8; q += 256) {
    int r = q >> 4;
    int e = e0 + r;
    if (e < E) ((uint4*)(X1 + (long long)e * H1))[q & 15] = ot[q];
  }
}

// ---------------- BN1 finalize, fold into W2, pack W2 into WMMA B layout ----------------
__global__ __launch_bounds__(256) void bn1_fold(
    const float* __restrict__ S1, const float* __restrict__ G1,
    const float* __restrict__ BE1, const float* __restrict__ W2,
    const float* __restrict__ B2, uint32_t* __restrict__ W2p,
    float* __restrict__ B2p, int E) {
  __shared__ float sc1[H1], sh1[H1];
  int t = threadIdx.x;
  if (t < H1) {
    float s = 0.f, s2 = 0.f;
    for (int b = 0; b < NBINS; ++b) {
      s  += S1[b * 2 * H1 + t];
      s2 += S1[b * 2 * H1 + H1 + t];
    }
    float inv = 1.f / (float)E;
    float mean = s * inv;
    float var = s2 * inv - mean * mean;
    float sc = G1[t] * rsqrtf(var + BN_EPS);
    sc1[t] = sc;
    sh1[t] = BE1[t] - mean * sc;
  }
  __syncthreads();
  if (t < H2) {
    float a = B2[t];
    for (int k = 0; k < H1; ++k) a += sh1[k] * W2[k * H2 + t];
    B2p[t] = a;
  }
  for (int idx = t; idx < K2_CHUNKS * 4 * 32 * 8; idx += 256) {
    int j = idx & 7, lane = (idx >> 3) & 31, nt = (idx >> 8) & 3, c = idx >> 10;
    int n = (nt << 4) + (lane & 15);
    int k = (c << 5) + ((lane >> 4) << 4) + (j << 1);
    H2U p;
    p.h[0] = (_Float16)(sc1[k] * W2[k * H2 + n]);
    p.h[1] = (_Float16)(sc1[k + 1] * W2[(k + 1) * H2 + n]);
    W2p[idx] = p.u;
  }
}

// ---------------- layer 2: [32 x 128] @ [128 x 64] WMMA + ReLU + stats ----------------
__global__ __launch_bounds__(256) void layer2(
    const _Float16* __restrict__ X1, const uint32_t* __restrict__ W2p,
    const float* __restrict__ B2p, _Float16* __restrict__ X2,
    float* __restrict__ S2, int E) {
  __shared__ _Float16 otile[32 * H2];
  const int tid = threadIdx.x, wave = tid >> 5, lane = tid & 31;
  const int nt = wave & 3, mt = wave >> 2;
  const int row = lane & 15, kbh = (lane >> 4) << 3;
  const int e0 = blockIdx.x * 32;
  int erow = e0 + (mt << 4) + row;
  int eclamp = (erow < E) ? erow : (E - 1);   // keep EXEC all-ones for WMMA
  const uint32_t* xrow = (const uint32_t*)X1 + (long long)eclamp * (H1 / 2);

  v8f acc = {};
  for (int c = 0; c < K2_CHUNKS; ++c) {
    AV a, b;
    const uint4* bq = (const uint4*)(W2p + ((((c << 2) + nt) << 5) + lane) * 8);
    b.q[0] = bq[0]; b.q[1] = bq[1];
    a.q[0] = *(const uint4*)(xrow + (c << 4) + kbh);
    a.q[1] = *(const uint4*)(xrow + (c << 4) + kbh + 4);
    acc = __builtin_amdgcn_wmma_f32_16x16x32_f16(false, a.v, false, b.v,
                                                 (short)0, acc, false, false);
  }

  const int n = (lane & 15) + (nt << 4);
  const int mb = (lane >> 4) << 3;
  const float bias = B2p[n];
  float s = 0.f, s2 = 0.f;
  for (int i = 0; i < 8; ++i) {
    int m = (mt << 4) + mb + i;
    float v = fmaxf(acc[i] + bias, 0.f);
    otile[m * H2 + n] = (_Float16)v;
    if (e0 + m < E) { s += v; s2 += v * v; }
  }
  s  += __shfl_xor(s, 16, 32);
  s2 += __shfl_xor(s2, 16, 32);
  if (lane < 16) {
    float* bin = S2 + (size_t)(blockIdx.x & (NBINS - 1)) * (2 * H2);
    atomicAdd(bin + n, s);
    atomicAdd(bin + H2 + n, s2);
  }
  __syncthreads();

  const uint4* ot = (const uint4*)otile;
  for (int q = tid; q < 32 * H2 / 8; q += 256) {
    int r = q >> 3;
    int e = e0 + r;
    if (e < E) ((uint4*)(X2 + (long long)e * H2))[q & 7] = ot[q];
  }
}

// ---------------- BN2 finalize, fold into W3 ----------------
__global__ __launch_bounds__(64) void bn2_fold(
    const float* __restrict__ S2, const float* __restrict__ G2,
    const float* __restrict__ BE2, const float* __restrict__ W3,
    const float* __restrict__ B3, float* __restrict__ W3p, int E) {
  __shared__ float tmp[H2];
  int t = threadIdx.x;
  float s = 0.f, s2 = 0.f;
  for (int b = 0; b < NBINS; ++b) {
    s  += S2[b * 2 * H2 + t];
    s2 += S2[b * 2 * H2 + H2 + t];
  }
  float inv = 1.f / (float)E;
  float mean = s * inv;
  float var = s2 * inv - mean * mean;
  float sc = G2[t] * rsqrtf(var + BN_EPS);
  float sh = BE2[t] - mean * sc;
  float w = W3[t];
  W3p[t] = sc * w;
  tmp[t] = sh * w;
  __syncthreads();
  if (t == 0) {
    float a = B3[0];
    for (int k = 0; k < H2; ++k) a += tmp[k];
    W3p[H2] = a;      // folded bias
  }
}

// ---------------- layer 3: per-edge 64-dot + tanh ----------------
__device__ __forceinline__ float dot8(uint4 u, const float* w) {
  H2U a, b, c, d;
  a.u = u.x; b.u = u.y; c.u = u.z; d.u = u.w;
  return (float)a.h[0] * w[0] + (float)a.h[1] * w[1] +
         (float)b.h[0] * w[2] + (float)b.h[1] * w[3] +
         (float)c.h[0] * w[4] + (float)c.h[1] * w[5] +
         (float)d.h[0] * w[6] + (float)d.h[1] * w[7];
}

__global__ __launch_bounds__(256) void layer3(
    const _Float16* __restrict__ X2, const float* __restrict__ W3p,
    float* __restrict__ OUT, int E) {
  __shared__ float w3s[H2];
  __shared__ float b3s;
  int tid = threadIdx.x;
  if (tid < H2) w3s[tid] = W3p[tid];
  if (tid == 0) b3s = W3p[H2];
  __syncthreads();
  int e = blockIdx.x * 256 + tid;
  if (e >= E) return;
  const uint4* xr = (const uint4*)(X2 + (long long)e * H2);
  float acc = b3s;
  for (int q = 0; q < H2 / 8; ++q) acc += dot8(xr[q], &w3s[q * 8]);
  OUT[e] = tanhf(acc);
}

// ---------------- host launch ----------------
extern "C" void kernel_launch(void* const* d_in, const int* in_sizes, int n_in,
                              void* d_out, int out_size, void* d_ws, size_t ws_size,
                              hipStream_t stream) {
  const float*     NE   = (const float*)d_in[0];
  const float*     GS   = (const float*)d_in[1];
  const float*     STEP = (const float*)d_in[2];
  const float*     W1   = (const float*)d_in[3];
  const float*     B1   = (const float*)d_in[4];
  const float*     G1   = (const float*)d_in[5];
  const float*     BE1  = (const float*)d_in[6];
  const float*     W2   = (const float*)d_in[7];
  const float*     B2   = (const float*)d_in[8];
  const float*     G2   = (const float*)d_in[9];
  const float*     BE2  = (const float*)d_in[10];
  const float*     W3   = (const float*)d_in[11];
  const float*     B3   = (const float*)d_in[12];
  const long long* EI   = (const long long*)d_in[13];
  float* OUT = (float*)d_out;
  const int E = in_sizes[1];
  (void)n_in; (void)out_size; (void)ws_size;

  char* ws = (char*)d_ws;
  size_t off = 0;
  auto take = [&](size_t bytes) -> void* {
    void* p = ws + off;
    off = (off + bytes + 255) & ~(size_t)255;
    return p;
  };
  _Float16* X1  = (_Float16*)take((size_t)E * H1 * sizeof(_Float16));
  _Float16* X2  = (_Float16*)take((size_t)E * H2 * sizeof(_Float16));
  uint32_t* B1p = (uint32_t*)take((size_t)K1_CHUNKS * 8 * 32 * 8 * 4);
  uint32_t* W2p = (uint32_t*)take((size_t)K2_CHUNKS * 4 * 32 * 8 * 4);
  float*    S1  = (float*)take((size_t)NBINS * 2 * H1 * 4);
  float*    S2  = (float*)take((size_t)NBINS * 2 * H2 * 4);
  float*    B2p = (float*)take((size_t)H2 * 4);
  float*    W3p = (float*)take((size_t)(H2 + 1) * 4);

  zero_stats<<<64, 256, 0, stream>>>(S1, S2);
  pack_w1<<<(K1_CHUNKS * 8 * 32 * 8 + 255) / 256, 256, 0, stream>>>(W1, B1p);
  layer1<<<(E + 31) / 32, 256, 0, stream>>>(NE, GS, STEP, W1, B1, EI, B1p, X1, S1, E);
  bn1_fold<<<1, 256, 0, stream>>>(S1, G1, BE1, W2, B2, W2p, B2p, E);
  layer2<<<(E + 31) / 32, 256, 0, stream>>>(X1, W2p, B2p, X2, S2, E);
  bn2_fold<<<1, 64, 0, stream>>>(S2, G2, BE2, W3, B3, W3p, E);
  layer3<<<(E + 255) / 256, 256, 0, stream>>>(X2, W3p, OUT, E);
}
